// MoE_44684839748081
// MI455X (gfx1250) — matrix operand order repaired and checked by reading, compiled, not run on hardware
//
#include <hip/hip_runtime.h>
#include <hip/hip_bf16.h>

// Problem dims (fixed by the reference)
#define NTOK   4096      // B*T
#define DMODEL 1024
#define DFF    512
#define NEXP   8

// LDS pitches (pad by 8 bf16 = 16B so row stride is 4 banks -> conflict-light)
#define XPITCH (DMODEL + 8)   // 1032
#define HPITCH (DFF + 8)      // 520

typedef __attribute__((ext_vector_type(16))) __bf16 v16bf;
typedef __attribute__((ext_vector_type(8)))  __bf16 bf16x8;
typedef __attribute__((ext_vector_type(8)))  float  v8f;

// Workspace layout (bytes); all regions >=32B aligned for v16bf loads
#define OFF_COUNTS 0
#define OFF_LIST   256
#define OFF_PLIST  (OFF_LIST  + (size_t)NEXP * NTOK * 4)         // 256 + 128K
#define OFF_WG     (OFF_PLIST + (size_t)NEXP * NTOK * 4)
#define OFF_WU     (OFF_WG + (size_t)NEXP * DFF * DMODEL * 2)
#define OFF_WD     (OFF_WU + (size_t)NEXP * DFF * DMODEL * 2)
#define OFF_C1     (OFF_WD + (size_t)NEXP * DMODEL * DFF * 2)

// Assemble a 16x32-bf16 A fragment (contiguous K order, matching B):
// lanes 0-15: row M=lane,   elems 0-15 = K(k0+0..15)
// lanes 16-31: row M=lane-16, elems 0-15 = K(k0+16..31)
__device__ __forceinline__ v16bf frag_cat(const __bf16* lo, const __bf16* hi) {
  bf16x8 a = *(const bf16x8*)lo;
  bf16x8 b = *(const bf16x8*)hi;
  v16bf r;
#pragma unroll
  for (int i = 0; i < 8; ++i) { r[i] = a[i]; r[i + 8] = b[i]; }
  return r;
}

__global__ void k_zero_counts(int* counts) {
  if (threadIdx.x < NEXP) counts[threadIdx.x] = 0;
}

// fp32 -> bf16 one-time weight conversion (streamed once; ~72MB total traffic)
__global__ __launch_bounds__(256) void k_convert(
    const float* __restrict__ wg, const float* __restrict__ wu,
    const float* __restrict__ wd,
    __bf16* __restrict__ wgb, __bf16* __restrict__ wub, __bf16* __restrict__ wdb) {
  const int total = NEXP * DFF * DMODEL;
  for (int i = blockIdx.x * blockDim.x + threadIdx.x; i < total;
       i += gridDim.x * blockDim.x) {
    wgb[i] = (__bf16)wg[i];
    wub[i] = (__bf16)wu[i];
    wdb[i] = (__bf16)wd[i];
  }
}

// Router: one wave32 per token. router_w staged in LDS (32KB).
__global__ __launch_bounds__(256) void k_router(
    const float* __restrict__ x, const float* __restrict__ rw,
    float* __restrict__ logits, int* __restrict__ counts,
    int* __restrict__ list, float* __restrict__ plist) {
  __shared__ float rw_s[NEXP * DMODEL];
  const int tid = threadIdx.x;
  for (int i = tid; i < NEXP * DMODEL; i += 256) rw_s[i] = rw[i];
  __syncthreads();

  const int wv = tid >> 5, lane = tid & 31;
  const int tok = blockIdx.x * 8 + wv;
  const float* xr = x + (size_t)tok * DMODEL;

  float acc[NEXP];
#pragma unroll
  for (int e = 0; e < NEXP; ++e) acc[e] = 0.0f;
  for (int j = 0; j < DMODEL / 32; ++j) {
    float xv = xr[lane + 32 * j];
#pragma unroll
    for (int e = 0; e < NEXP; ++e) acc[e] += xv * rw_s[e * DMODEL + lane + 32 * j];
  }
#pragma unroll
  for (int e = 0; e < NEXP; ++e)
    for (int off = 16; off > 0; off >>= 1)
      acc[e] += __shfl_xor(acc[e], off, 32);

  if (lane == 0) {
#pragma unroll
    for (int e = 0; e < NEXP; ++e) logits[(size_t)tok * NEXP + e] = acc[e];
    // top-2 with jax.lax.top_k tie-breaking (lowest index first)
    int s0 = 0; float m0 = acc[0];
#pragma unroll
    for (int e = 1; e < NEXP; ++e) if (acc[e] > m0) { m0 = acc[e]; s0 = e; }
    int s1 = 0; float m1 = -3.402823466e38f;
#pragma unroll
    for (int e = 0; e < NEXP; ++e)
      if (e != s0 && acc[e] > m1) { m1 = acc[e]; s1 = e; }
    float t  = __expf(m1 - m0);
    float p0 = 1.0f / (1.0f + t);
    float p1 = t / (1.0f + t);
    int pos0 = atomicAdd(&counts[s0], 1);
    list[s0 * NTOK + pos0]  = (tok << 1);
    plist[s0 * NTOK + pos0] = p0;
    int pos1 = atomicAdd(&counts[s1], 1);
    list[s1 * NTOK + pos1]  = (tok << 1) | 1;
    plist[s1 * NTOK + pos1] = p1;
  }
}

// Grouped expert FFN: block = 16 gathered tokens of one expert, 8 waves.
// Phase 1: H[16,512] = silu(X Wg^T) * (X Wu^T)   (bf16 WMMA, f32 acc)
// Phase 2: OUT[16,1024] = (H Wd^T) * prob        (bf16 WMMA, f32 acc)
__global__ __launch_bounds__(256) void k_ffn(
    const float*  __restrict__ x,
    const __bf16* __restrict__ wgb,
    const __bf16* __restrict__ wub,
    const __bf16* __restrict__ wdb,
    const int*    __restrict__ counts,
    const int*    __restrict__ list,
    const float*  __restrict__ plist,
    float* __restrict__ out0,   // slot-0 rows -> d_out directly
    float* __restrict__ c1) {   // slot-1 rows -> scratch
  __shared__ __bf16 x_lds[16 * XPITCH];   // 33.0 KB
  __shared__ __bf16 h_lds[16 * HPITCH];   // 16.6 KB
  __shared__ int    tok_s[16];
  __shared__ int    slot_s[16];
  __shared__ float  prob_s[16];

  const int bid  = blockIdx.x;
  const int e    = bid >> 8;     // 256 tiles reserved per expert
  const int tile = bid & 255;
  const int cnt  = counts[e];
  if (tile * 16 >= cnt) return;

  const int tid = threadIdx.x;
  if (tid < 16) {
    int idx = tile * 16 + tid;
    if (idx < cnt) {
      int entry    = list[e * NTOK + idx];
      tok_s[tid]   = entry >> 1;
      slot_s[tid]  = entry & 1;
      prob_s[tid]  = plist[e * NTOK + idx];
    } else {
      tok_s[tid] = -1; slot_s[tid] = 0; prob_s[tid] = 0.0f;
    }
  }
  __syncthreads();

  // Gather + convert the 16 token rows of x into LDS (bf16)
  for (int i = tid; i < 16 * DMODEL; i += 256) {
    int row = i >> 10;
    int col = i & (DMODEL - 1);
    int t = tok_s[row];
    int g = (t < 0) ? 0 : t;
    x_lds[row * XPITCH + col] = (__bf16)x[(size_t)g * DMODEL + col];
  }
  __syncthreads();

  const int wv   = tid >> 5;
  const int lane = tid & 31;
  const int hlf  = lane >> 4;   // lane half
  const int ln   = lane & 15;   // row (A) / col (B,D) index
  const int klo  = hlf * 16;    // A-frag K offset (contiguous 16 elems)
  const int kb16 = hlf * 16;    // B-frag K offset (contiguous 16 elems)
  const int mb   = hlf * 8;     // D-matrix row base

  // ---- Phase 1: 32 i-tiles across 8 waves ----
  for (int it = wv; it < DFF / 16; it += 8) {
    v8f accg = {};
    v8f accu = {};
    const __bf16* bg_row = wgb + ((size_t)e * DFF + it * 16 + ln) * DMODEL + kb16;
    const __bf16* bu_row = wub + ((size_t)e * DFF + it * 16 + ln) * DMODEL + kb16;
    for (int ks = 0; ks < DMODEL / 32; ++ks) {
      const __bf16* ap = &x_lds[ln * XPITCH + ks * 32 + klo];
      v16bf a  = frag_cat(ap, ap + 8);
      v16bf bg = *(const v16bf*)(bg_row + ks * 32);
      v16bf bu = *(const v16bf*)(bu_row + ks * 32);
      accg = __builtin_amdgcn_wmma_f32_16x16x32_bf16(false, a, false, bg,
                                                     (short)0, accg, false, false);
      accu = __builtin_amdgcn_wmma_f32_16x16x32_bf16(false, a, false, bu,
                                                     (short)0, accu, false, false);
    }
#pragma unroll
    for (int r = 0; r < 8; ++r) {
      float g = accg[r];
      float u = accu[r];
      float h = g * (1.0f / (1.0f + __expf(-g))) * u;   // silu(g)*u
      h_lds[(r + mb) * HPITCH + it * 16 + ln] = (__bf16)h;
    }
  }
  __syncthreads();

  // ---- Phase 2: 64 d-tiles across 8 waves ----
  for (int nt = wv; nt < DMODEL / 16; nt += 8) {
    v8f acc = {};
    const __bf16* bd_row = wdb + ((size_t)e * DMODEL + nt * 16 + ln) * DFF + kb16;
    for (int ks = 0; ks < DFF / 32; ++ks) {
      const __bf16* ap = &h_lds[ln * HPITCH + ks * 32 + klo];
      v16bf a = frag_cat(ap, ap + 8);
      v16bf b = *(const v16bf*)(bd_row + ks * 32);
      acc = __builtin_amdgcn_wmma_f32_16x16x32_bf16(false, a, false, b,
                                                    (short)0, acc, false, false);
    }
#pragma unroll
    for (int r = 0; r < 8; ++r) {
      int m = r + mb;
      int t = tok_s[m];
      if (t >= 0) {
        float v = acc[r] * prob_s[m];
        float* dst = (slot_s[m] ? c1 : out0) + (size_t)t * DMODEL + nt * 16 + ln;
        *dst = v;
      }
    }
  }
}

__global__ __launch_bounds__(256) void k_combine(float* __restrict__ out,
                                                 const float* __restrict__ c1) {
  const int total = NTOK * DMODEL;
  for (int i = blockIdx.x * blockDim.x + threadIdx.x; i < total;
       i += gridDim.x * blockDim.x)
    out[i] += c1[i];
}

extern "C" void kernel_launch(void* const* d_in, const int* in_sizes, int n_in,
                              void* d_out, int out_size, void* d_ws, size_t ws_size,
                              hipStream_t stream) {
  const float* x  = (const float*)d_in[0];   // [B,T,D]
  const float* rw = (const float*)d_in[1];   // [E,D]
  const float* wg = (const float*)d_in[2];   // [E,I,D]
  const float* wu = (const float*)d_in[3];   // [E,I,D]
  const float* wd = (const float*)d_in[4];   // [E,D,I]

  float* out    = (float*)d_out;                      // [NTOK, D]
  float* logits = out + (size_t)NTOK * DMODEL;        // [NTOK, E]

  char* ws = (char*)d_ws;
  int*    counts = (int*)   (ws + OFF_COUNTS);
  int*    list   = (int*)   (ws + OFF_LIST);
  float*  plist  = (float*) (ws + OFF_PLIST);
  __bf16* wgb    = (__bf16*)(ws + OFF_WG);
  __bf16* wub    = (__bf16*)(ws + OFF_WU);
  __bf16* wdb    = (__bf16*)(ws + OFF_WD);
  float*  c1     = (float*) (ws + OFF_C1);

  k_zero_counts<<<1, 32, 0, stream>>>(counts);
  k_convert<<<2048, 256, 0, stream>>>(wg, wu, wd, wgb, wub, wdb);
  k_router<<<NTOK / 8, 256, 0, stream>>>(x, rw, logits, counts, list, plist);
  k_ffn<<<NEXP * 256, 256, 0, stream>>>(x, wgb, wub, wdb, counts, list, plist,
                                        out, c1);
  k_combine<<<2048, 256, 0, stream>>>(out, c1);
}
